// LinearAttention_5076651343921
// MI455X (gfx1250) — matrix operand order repaired and checked by reading, not compiled, not run on hardware
//
#include <hip/hip_runtime.h>

typedef __attribute__((ext_vector_type(16))) _Float16 v16h;
typedef __attribute__((ext_vector_type(2)))  _Float16 v2h;
typedef __attribute__((ext_vector_type(8)))  float    v8f;

#define S_LEN 2048
#define DH    64
#define NBH   64          // B*H = 4*16
#define SEGS  8           // s-segments for KV accumulation
#define SROWS (S_LEN / SEGS)
#define CH    32          // s-chunk per WMMA K step
#define ANG_SCALE 7.66990393942820614e-4f   // (pi/2)/2048
#define INV_TEMP  0.125f

__device__ __forceinline__ float nonneg(float x) {
  return (x < 0.0f) ? __expf(x) : (x + 1.0f);
}

__global__ void zero_kernel(float* __restrict__ p, int n) {
  int i = blockIdx.x * blockDim.x + threadIdx.x;
  if (i < n) p[i] = 0.0f;
}

// ---------------------------------------------------------------------------
// Phase 1: KV_cos/KV_sin[64x64] = sum_s feat(k)[s]*{cos,sin}(s) (x) v[s]
// grid: (NBH, SEGS), block: 128 (4 waves).
// LDS holds data pre-swizzled into WMMA fragment order:
//   A frag  [wave][lane][i], i = (kk>>4)*8 | (kk&7), lane = (kk&8?16:0)+m
//   B frag  [ntile][lane][i], i = (kk>>4)*8 | (kk&7), lane = (kk&8?16:0)+n
// so fragment reads are 2x ds_load_b128 per lane, writes are b32 (row pairs).
// ---------------------------------------------------------------------------
__global__ __launch_bounds__(128) void cosformer_kv(const float* __restrict__ K,
                                                    const float* __restrict__ V,
                                                    float* __restrict__ kvws) {
  const int bh   = blockIdx.x;
  const int seg  = blockIdx.y;
  const int tid  = threadIdx.x;
  const int lane = tid & 31;
  const int wave = tid >> 5;

  __shared__ __align__(32) _Float16 AfC[4][32][16];   // feat(k)*cos, A-frag order
  __shared__ __align__(32) _Float16 AfS[4][32][16];   // feat(k)*sin, A-frag order
  __shared__ __align__(32) _Float16 BfV[4][32][16];   // v,           B-frag order

  const float* kb = K + (size_t)bh * S_LEN * DH;
  const float* vb = V + (size_t)bh * S_LEN * DH;

  v8f zero = {};
  v8f accc[4], accs[4];
#pragma unroll
  for (int t = 0; t < 4; ++t) { accc[t] = zero; accs[t] = zero; }

  // Writer mapping: thread owns s-rows (kk0, kk0+1) x 8 cols.
  const int kk0    = (tid >> 3) << 1;            // even row in chunk, 0..30
  const int col0   = (tid & 7) << 3;             // 0,8,...,56
  const int iA     = ((kk0 >> 4) << 3) | (kk0 & 7);   // even
  const int lAbase = (kk0 & 8) ? 16 : 0;
  const int iB     = ((kk0 >> 4) << 3) | (kk0 & 7);   // even
  const int lBbase = (kk0 & 8) ? 16 : 0;

  for (int s0 = seg * SROWS; s0 < (seg + 1) * SROWS; s0 += CH) {
    const int sA = s0 + kk0;
    const int sB = sA + 1;
    const float cr0 = __cosf(ANG_SCALE * (float)sA);
    const float sr0 = __sinf(ANG_SCALE * (float)sA);
    const float cr1 = __cosf(ANG_SCALE * (float)sB);
    const float sr1 = __sinf(ANG_SCALE * (float)sB);

    const float4* kp0 = reinterpret_cast<const float4*>(kb + (size_t)sA * DH + col0);
    const float4* kp1 = reinterpret_cast<const float4*>(kb + (size_t)sB * DH + col0);
    const float4* vp0 = reinterpret_cast<const float4*>(vb + (size_t)sA * DH + col0);
    const float4* vp1 = reinterpret_cast<const float4*>(vb + (size_t)sB * DH + col0);
    float k0[8], k1[8], v0[8], v1[8];
#pragma unroll
    for (int t = 0; t < 2; ++t) {
      float4 a = kp0[t]; k0[t*4+0]=a.x; k0[t*4+1]=a.y; k0[t*4+2]=a.z; k0[t*4+3]=a.w;
      float4 b = kp1[t]; k1[t*4+0]=b.x; k1[t*4+1]=b.y; k1[t*4+2]=b.z; k1[t*4+3]=b.w;
      float4 c = vp0[t]; v0[t*4+0]=c.x; v0[t*4+1]=c.y; v0[t*4+2]=c.z; v0[t*4+3]=c.w;
      float4 d = vp1[t]; v1[t*4+0]=d.x; v1[t*4+1]=d.y; v1[t*4+2]=d.z; v1[t*4+3]=d.w;
    }
    // Speculative prefetch of next chunk (silently dropped when out of range).
    __builtin_prefetch(kb + (size_t)(sA + CH) * DH + col0, 0, 0);
    __builtin_prefetch(vb + (size_t)(sA + CH) * DH + col0, 0, 0);

#pragma unroll
    for (int j = 0; j < 8; ++j) {
      const int col = col0 + j;
      const int w   = col >> 4;
      const int mn  = col & 15;
      const float f0 = nonneg(k0[j]);
      const float f1 = nonneg(k1[j]);
      v2h pc; pc[0] = (_Float16)(f0 * cr0); pc[1] = (_Float16)(f1 * cr1);
      v2h ps; ps[0] = (_Float16)(f0 * sr0); ps[1] = (_Float16)(f1 * sr1);
      v2h pv; pv[0] = (_Float16)v0[j];      pv[1] = (_Float16)v1[j];
      *(v2h*)&AfC[w][lAbase + mn][iA] = pc;
      *(v2h*)&AfS[w][lAbase + mn][iA] = ps;
      *(v2h*)&BfV[w][lBbase + mn][iB] = pv;
    }
    __syncthreads();

    const v16h Ac = *(const v16h*)&AfC[wave][lane][0];
    const v16h As = *(const v16h*)&AfS[wave][lane][0];
#pragma unroll
    for (int nt = 0; nt < 4; ++nt) {
      const v16h Bv = *(const v16h*)&BfV[nt][lane][0];
      accc[nt] = __builtin_amdgcn_wmma_f32_16x16x32_f16(false, Ac, false, Bv,
                                                        (short)0, accc[nt], false, false);
      accs[nt] = __builtin_amdgcn_wmma_f32_16x16x32_f16(false, As, false, Bv,
                                                        (short)0, accs[nt], false, false);
    }
    __syncthreads();
  }

  float* kvc = kvws + (size_t)bh * 2 * DH * DH;
  float* kvs = kvc + DH * DH;
  const int mrow0 = (wave << 4) + ((lane < 16) ? 0 : 8);
  const int nlo   = lane & 15;
#pragma unroll
  for (int nt = 0; nt < 4; ++nt) {
    const int n = (nt << 4) + nlo;
#pragma unroll
    for (int r = 0; r < 8; ++r) {
      const int mm = mrow0 + r;
      atomicAdd(&kvc[mm * DH + n], accc[nt][r]);
      atomicAdd(&kvs[mm * DH + n], accs[nt][r]);
    }
  }
}

// ---------------------------------------------------------------------------
// Phase 2: out[s,:] = mask[s] * (q_cos[s,:]@KV_cos + q_sin[s,:]@KV_sin) / T
// grid: (NBH, S_LEN/64), block: 128 (4 waves, each owns 16 q-rows).
// Same fragment-order LDS staging as phase 1.
// ---------------------------------------------------------------------------
__global__ __launch_bounds__(128) void cosformer_out(const float* __restrict__ Q,
                                                     const float* __restrict__ kvws,
                                                     const int* __restrict__ tmask,
                                                     float* __restrict__ out) {
  const int bh   = blockIdx.x;
  const int b    = bh >> 4;          // H = 16
  const int sc   = blockIdx.y;       // 64-row chunk
  const int tid  = threadIdx.x;
  const int lane = tid & 31;
  const int wave = tid >> 5;

  __shared__ __align__(32) _Float16 qAc[4][2][32][16];  // [wave][kchunk][lane][i]
  __shared__ __align__(32) _Float16 qAs[4][2][32][16];
  __shared__ __align__(32) _Float16 kvBc[2][4][32][16]; // [kchunk][ntile][lane][i]
  __shared__ __align__(32) _Float16 kvBs[2][4][32][16];

  const float* kvc = kvws + (size_t)bh * 2 * DH * DH;
  const float* kvs = kvc + DH * DH;

  // Stage KV into B-fragment order: thread owns column n, 16 kdim row-pairs.
  {
    const int n   = tid & 63;
    const int nt  = n >> 4;
    const int nl  = n & 15;
    const int rpb = tid >> 6;        // 0..1
#pragma unroll
    for (int p = 0; p < 16; ++p) {
      const int kd0   = (((rpb << 4) + p) << 1); // even, 0..62
      const int chunk = kd0 >> 5;
      const int kk0   = kd0 & 31;
      const int laneB = ((kk0 & 8) ? 16 : 0) + nl;
      const int i0    = ((kk0 >> 4) << 3) | (kk0 & 7);
      v2h pc; pc[0] = (_Float16)kvc[kd0 * DH + n]; pc[1] = (_Float16)kvc[(kd0 + 1) * DH + n];
      v2h ps; ps[0] = (_Float16)kvs[kd0 * DH + n]; ps[1] = (_Float16)kvs[(kd0 + 1) * DH + n];
      *(v2h*)&kvBc[chunk][nt][laneB][i0] = pc;
      *(v2h*)&kvBs[chunk][nt][laneB][i0] = ps;
    }
  }

  const int s0 = sc * 64;
  // Stage q features into A-fragment order: thread owns one row, 32 cols.
  {
    const int row  = tid >> 1;          // 0..63
    const int col0 = (tid & 1) << 5;    // 0 or 32
    const int s    = s0 + row;
    float cr = __cosf(ANG_SCALE * (float)s);
    float sr = __sinf(ANG_SCALE * (float)s);
    const float mv = tmask[b * S_LEN + s] ? 1.0f : 0.0f;
    cr *= mv; sr *= mv;
    const int w = row >> 4;
    const int m = row & 15;
    const float4* qp = reinterpret_cast<const float4*>(Q + ((size_t)bh * S_LEN + s) * DH + col0);
    float qv[32];
#pragma unroll
    for (int t = 0; t < 8; ++t) {
      float4 a = qp[t];
      qv[t*4+0] = a.x; qv[t*4+1] = a.y; qv[t*4+2] = a.z; qv[t*4+3] = a.w;
    }
#pragma unroll
    for (int jj = 0; jj < 16; ++jj) {            // col pairs (even kdim)
      const int kd0   = col0 + (jj << 1);
      const int chunk = kd0 >> 5;
      const int kk0   = kd0 & 31;
      const int laneA = ((kk0 & 8) ? 16 : 0) + m;
      const int i0    = ((kk0 >> 4) << 3) | (kk0 & 7);
      const float f0 = nonneg(qv[jj * 2]);
      const float f1 = nonneg(qv[jj * 2 + 1]);
      v2h pc; pc[0] = (_Float16)(f0 * cr); pc[1] = (_Float16)(f1 * cr);
      v2h ps; ps[0] = (_Float16)(f0 * sr); ps[1] = (_Float16)(f1 * sr);
      *(v2h*)&qAc[w][chunk][laneA][i0] = pc;
      *(v2h*)&qAs[w][chunk][laneA][i0] = ps;
    }
  }
  __syncthreads();

  v8f zero = {};
  v8f acc[4];
#pragma unroll
  for (int t = 0; t < 4; ++t) acc[t] = zero;

#pragma unroll
  for (int c2 = 0; c2 < 2; ++c2) {               // K = 64 in two chunks of 32
    const v16h Ac = *(const v16h*)&qAc[wave][c2][lane][0];
    const v16h As = *(const v16h*)&qAs[wave][c2][lane][0];
#pragma unroll
    for (int nt = 0; nt < 4; ++nt) {
      const v16h Bc = *(const v16h*)&kvBc[c2][nt][lane][0];
      const v16h Bs = *(const v16h*)&kvBs[c2][nt][lane][0];
      acc[nt] = __builtin_amdgcn_wmma_f32_16x16x32_f16(false, Ac, false, Bc,
                                                       (short)0, acc[nt], false, false);
      acc[nt] = __builtin_amdgcn_wmma_f32_16x16x32_f16(false, As, false, Bs,
                                                       (short)0, acc[nt], false, false);
    }
  }

  const int mrow0 = (wave << 4) + ((lane < 16) ? 0 : 8);
  const int nl    = lane & 15;
#pragma unroll
  for (int nt = 0; nt < 4; ++nt) {
    const int n = (nt << 4) + nl;
#pragma unroll
    for (int r = 0; r < 8; ++r) {
      const int srow = s0 + mrow0 + r;
      out[((size_t)bh * S_LEN + srow) * DH + n] = acc[nt][r] * INV_TEMP;
    }
  }
}

extern "C" void kernel_launch(void* const* d_in, const int* in_sizes, int n_in,
                              void* d_out, int out_size, void* d_ws, size_t ws_size,
                              hipStream_t stream) {
  const float* q  = (const float*)d_in[0];
  const float* k  = (const float*)d_in[1];
  const float* v  = (const float*)d_in[2];
  const int*   tm = (const int*)d_in[3];
  float* out  = (float*)d_out;
  float* kvws = (float*)d_ws;              // NBH * 2 * 64*64 floats = 2 MB

  const int nkv = NBH * 2 * DH * DH;
  zero_kernel<<<(nkv + 255) / 256, 256, 0, stream>>>(kvws, nkv);

  dim3 g1(NBH, SEGS);
  cosformer_kv<<<g1, 128, 0, stream>>>(k, v, kvws);

  dim3 g2(NBH, S_LEN / 64);
  cosformer_out<<<g2, 128, 0, stream>>>(q, kvws, tm, out);
}
